// AggregatorV1_74182675137087
// MI455X (gfx1250) — compile-verified, matrix operand order checked by reading, not run on hardware
//
#include <hip/hip_runtime.h>
#include <hip/hip_bf16.h>
#include <math.h>

// ---------------- CDNA5 WMMA types ----------------
typedef __attribute__((ext_vector_type(16))) __bf16 v16bf;
typedef __attribute__((ext_vector_type(8)))  float  v8f;

union BF16Tile {
  v16bf v;
  uint4 q[2];
  __bf16 h[16];
};

__device__ inline __bf16 f2bf(float f) {
  unsigned u = __builtin_bit_cast(unsigned, f);
  u += 0x7FFFu + ((u >> 16) & 1u);                    // round-to-nearest-even
  unsigned short s = (unsigned short)(u >> 16);
  return __builtin_bit_cast(__bf16, s);
}
__device__ inline float sigm(float x) { return 1.0f / (1.0f + expf(-x)); }

// Problem constants (fixed by setup_inputs)
constexpr int B_  = 8;
constexpr int T_  = 1024;
constexpr int F_  = 512;
constexpr int H_  = 512;
constexpr int FF_ = 512;
constexpr int CS_ = 32;
constexpr int K_  = 32;    // T/CS
constexpr int HN_ = 8;     // heads
constexpr int HD_ = 64;    // head dim
constexpr int G4_ = 2048;  // 4*H

// ---------------- generic batched WMMA GEMM ----------------
// C[b1,b2][m,n] = alpha * sum_k A[m,k] * B(n,k) + bias[n] (+bias2[n])
// A: bf16, row-major, element (m,k) at A + m*lda + k            (always contiguous K)
// B: bf16, element (n,k) at B + n*ldbN + k*ldbK                 (VECB => ldbK==1)
// C: fp32 at C + m*ldc + n ; optional bf16 mirror Cbf (same layout)
template <bool VECB>
__global__ __launch_bounds__(256)
void wmma_gemm_kernel(const __bf16* __restrict__ A, int lda, long sA1, long sA2,
                      const __bf16* __restrict__ B, int ldbN, int ldbK, long sB1, long sB2,
                      float* __restrict__ C, int ldc, long sC1, long sC2,
                      __bf16* __restrict__ Cbf,
                      const float* __restrict__ bias, const float* __restrict__ bias2,
                      int M, int N, int Ktot, float alpha) {
  const int wave = blockIdx.x * (blockDim.x >> 5) + (threadIdx.x >> 5);
  const int ntn = N >> 4;
  const int ntm = (M + 15) >> 4;
  if (wave >= ntm * ntn) return;                 // wave-uniform: EXEC stays full
  const int tm = wave / ntn, tn = wave % ntn;

  A += (long)blockIdx.y * sA1 + (long)blockIdx.z * sA2;
  B += (long)blockIdx.y * sB1 + (long)blockIdx.z * sB2;
  C += (long)blockIdx.y * sC1 + (long)blockIdx.z * sC2;
  if (Cbf) Cbf += (long)blockIdx.y * sC1 + (long)blockIdx.z * sC2;

  const int lane = threadIdx.x & 31;
  const int l15 = lane & 15;
  const int hi = (lane >> 4) & 1;
  const int koff = hi ? 8 : 0;
  const int arow = tm * 16 + l15;
  const int bcol = tn * 16 + l15;
  const bool arow_ok = arow < M;

  v8f acc = {};
  for (int k0 = 0; k0 < Ktot; k0 += 32) {
    BF16Tile a, b;
    if (arow_ok) {
      const __bf16* pa = A + (long)arow * lda + k0 + koff;
      a.q[0] = *(const uint4*)(pa);
      a.q[1] = *(const uint4*)(pa + 16);
    } else {
      a.q[0] = uint4{0, 0, 0, 0};
      a.q[1] = uint4{0, 0, 0, 0};
    }
    if (VECB) {
      const __bf16* pb = B + (long)bcol * ldbN + k0 + koff;
      b.q[0] = *(const uint4*)(pb);
      b.q[1] = *(const uint4*)(pb + 16);
      __builtin_prefetch((const void*)(pb + 64), 0, 1);   // global_prefetch_b8
    } else {
#pragma unroll
      for (int j = 0; j < 8; ++j) {
        b.h[j]     = B[(long)bcol * ldbN + (long)(k0 + koff + j) * ldbK];
        b.h[j + 8] = B[(long)bcol * ldbN + (long)(k0 + koff + 16 + j) * ldbK];
      }
    }
    acc = __builtin_amdgcn_wmma_f32_16x16x32_bf16(false, a.v, false, b.v,
                                                  (short)0, acc, false, false);
  }

  float bval = bias ? bias[bcol] : 0.0f;
  if (bias2) bval += bias2[bcol];
#pragma unroll
  for (int r = 0; r < 8; ++r) {
    int row = tm * 16 + r + hi * 8;             // ISA C layout: lanes16-31 -> M=8+r
    if (row < M) {
      float vv = acc[r] * alpha + bval;
      C[(long)row * ldc + bcol] = vv;
      if (Cbf) Cbf[(long)row * ldc + bcol] = f2bf(vv);
    }
  }
}

// ---------------- LSTM recurrent scan (one layer, one workgroup) ----------------
// xg: [B,T,4H] fp32 precomputed x@Wih^T + bih + bhh ; Whh: [4H,H] bf16
// LDS: c[8*512] f32 | gates[8*2048] f32 | h_bf[16*512] bf16 (rows 8..15 = 0)
__global__ __launch_bounds__(1024)
void lstm_scan_kernel(const float* __restrict__ xg, const __bf16* __restrict__ Whh,
                      float* __restrict__ hs, __bf16* __restrict__ hs_bf) {
  extern __shared__ char smem[];
  float*  sc  = (float*)smem;                    // 16 KB
  float*  sg  = sc + B_ * H_;                    // 64 KB
  __bf16* shb = (__bf16*)(sg + B_ * G4_);        // 16 KB

  const int tid = threadIdx.x;
  for (int i = tid; i < B_ * H_; i += blockDim.x) sc[i] = 0.0f;
  for (int i = tid; i < 16 * H_; i += blockDim.x) shb[i] = f2bf(0.0f);
  __syncthreads();

  const int wave = tid >> 5, lane = tid & 31;
  const int l15 = lane & 15, hi = (lane >> 4) & 1, koff = hi ? 8 : 0;

  for (int t = 0; t < T_; ++t) {
    // gate pre-activations: g = h @ Whh^T + xg[:,t,:]
    for (int nt = wave; nt < G4_ / 16; nt += 32) {      // 4 N-tiles per wave
      const int bcol = nt * 16 + l15;
      v8f acc = {};
      for (int k0 = 0; k0 < H_; k0 += 32) {
        BF16Tile a, b;
        const __bf16* pa = shb + l15 * H_ + k0 + koff;          // ds_load_b128
        a.q[0] = *(const uint4*)(pa);
        a.q[1] = *(const uint4*)(pa + 16);
        const __bf16* pb = Whh + (long)bcol * H_ + k0 + koff;   // L2-resident
        b.q[0] = *(const uint4*)(pb);
        b.q[1] = *(const uint4*)(pb + 16);
        acc = __builtin_amdgcn_wmma_f32_16x16x32_bf16(false, a.v, false, b.v,
                                                      (short)0, acc, false, false);
      }
      if (hi == 0) {                                     // rows 0..7 = real batches
#pragma unroll
        for (int r = 0; r < 8; ++r)
          sg[r * G4_ + bcol] = acc[r] + xg[((long)r * T_ + t) * G4_ + bcol];
      }
    }
    __syncthreads();
    // elementwise gates (torch order i,f,g,o), fp32
    for (int i = tid; i < B_ * H_; i += blockDim.x) {
      int m = i >> 9, j = i & (H_ - 1);
      float gi = sg[m * G4_ + j];
      float gf = sg[m * G4_ + H_ + j];
      float gg = sg[m * G4_ + 2 * H_ + j];
      float go = sg[m * G4_ + 3 * H_ + j];
      float c = sigm(gf) * sc[i] + sigm(gi) * tanhf(gg);
      float h = sigm(go) * tanhf(c);
      sc[i] = c;
      shb[m * H_ + j] = f2bf(h);
      long o = ((long)m * T_ + t) * H_ + j;
      hs[o] = h;
      hs_bf[o] = f2bf(h);
    }
    __syncthreads();
  }
}

// ---------------- small fp32 kernels ----------------
__global__ void f2bf_kernel(const float* __restrict__ in, __bf16* __restrict__ out, long n) {
  long i = (long)blockIdx.x * blockDim.x + threadIdx.x;
  if (i < n) out[i] = f2bf(in[i]);
}

__global__ void layernorm_kernel(const float* __restrict__ x, const float* __restrict__ g,
                                 const float* __restrict__ b, float* __restrict__ y,
                                 __bf16* __restrict__ ybf, int rows) {
  int wave = blockIdx.x * (blockDim.x >> 5) + (threadIdx.x >> 5);
  if (wave >= rows) return;
  int lane = threadIdx.x & 31;
  const float* xr = x + (long)wave * H_;
  float s = 0.0f, s2 = 0.0f;
  for (int j = lane; j < H_; j += 32) { float v = xr[j]; s += v; s2 += v * v; }
  for (int m = 16; m; m >>= 1) { s += __shfl_xor(s, m, 32); s2 += __shfl_xor(s2, m, 32); }
  float mu = s / H_;
  float inv = rsqrtf(s2 / H_ - mu * mu + 1e-5f);
  for (int j = lane; j < H_; j += 32) {
    float v = (xr[j] - mu) * inv * g[j] + b[j];
    y[(long)wave * H_ + j] = v;
    ybf[(long)wave * H_ + j] = f2bf(v);
  }
}

__global__ void chunksum_kernel(const float* __restrict__ normed, __bf16* __restrict__ csum) {
  int row = blockIdx.x;           // b*K + k
  int j = threadIdx.x;            // 0..511
  float s = 0.0f;
  const float* p = normed + (long)row * CS_ * H_ + j;
  for (int i = 0; i < CS_; ++i) s += p[(long)i * H_];
  csum[(long)row * H_ + j] = f2bf(s);
}

__global__ void eof_kernel(const float* __restrict__ dp, const float* __restrict__ noise,
                           const unsigned char* __restrict__ mask, float* __restrict__ eofp) {
  long i = (long)blockIdx.x * blockDim.x + threadIdx.x;
  if (i >= (long)B_ * K_ * T_) return;
  int t = (int)(i % T_);
  long bk = i / T_;
  int k = (int)(bk % K_);
  int b = (int)(bk / K_);
  float e = (t < (k + 1) * CS_) ? 0.0f : sigm(dp[i] + noise[i]);
  if (!mask[(long)b * T_ + t]) e = 1.0f;
  eofp[i] = e;
}

__global__ void futctx_kernel(const float* __restrict__ eofp, float* __restrict__ p_ws,
                              float* __restrict__ p_out) {
  long i = (long)blockIdx.x * blockDim.x + threadIdx.x;
  if (i >= (long)B_ * K_ * T_) return;
  int t = (int)(i % T_);
  float v = (t == 0) ? 1.0f : 1.0f - eofp[i - 1];
  p_ws[i] = v;
  p_out[i] = v;
}

// softmax over t with expected-context weighting: attn = exp(s-max)*p / sum(exp(s-max)*p)
__global__ void softmax_w_kernel(const float* __restrict__ scores, const float* __restrict__ p,
                                 __bf16* __restrict__ attn_bf) {
  int wave = blockIdx.x * (blockDim.x >> 5) + (threadIdx.x >> 5);  // (b*HN + h)*K + k
  if (wave >= B_ * HN_ * K_) return;
  int lane = threadIdx.x & 31;
  int k = wave % K_;
  int b = wave / (HN_ * K_);
  const float* sr = scores + (long)wave * T_;
  const float* pr = p + ((long)b * K_ + k) * T_;
  float mx = -3.4e38f;
  for (int t = lane; t < T_; t += 32) mx = fmaxf(mx, sr[t]);
  for (int m = 16; m; m >>= 1) mx = fmaxf(mx, __shfl_xor(mx, m, 32));
  float sum = 0.0f;
  for (int t = lane; t < T_; t += 32) sum += expf(sr[t] - mx) * pr[t];
  for (int m = 16; m; m >>= 1) sum += __shfl_xor(sum, m, 32);
  float inv = 1.0f / sum;
  for (int t = lane; t < T_; t += 32)
    attn_bf[(long)wave * T_ + t] = f2bf(expf(sr[t] - mx) * pr[t] * inv);
}

// ---------------- host ----------------
static void launch_gemm(hipStream_t st,
                        const __bf16* A, int lda, long sA1, long sA2,
                        const __bf16* B, int ldbN, int ldbK, long sB1, long sB2,
                        float* C, int ldc, long sC1, long sC2, __bf16* Cbf,
                        const float* bias, const float* bias2,
                        int M, int N, int Ktot, float alpha, int b1, int b2) {
  int waves = ((M + 15) / 16) * (N / 16);
  dim3 grid((waves + 7) / 8, b1, b2), blk(256);
  if (ldbK == 1)
    wmma_gemm_kernel<true><<<grid, blk, 0, st>>>(A, lda, sA1, sA2, B, ldbN, ldbK, sB1, sB2,
                                                 C, ldc, sC1, sC2, Cbf, bias, bias2, M, N, Ktot, alpha);
  else
    wmma_gemm_kernel<false><<<grid, blk, 0, st>>>(A, lda, sA1, sA2, B, ldbN, ldbK, sB1, sB2,
                                                  C, ldc, sC1, sC2, Cbf, bias, bias2, M, N, Ktot, alpha);
}

extern "C" void kernel_launch(void* const* d_in, const int* in_sizes, int n_in,
                              void* d_out, int out_size, void* d_ws, size_t ws_size,
                              hipStream_t stream) {
  (void)in_sizes; (void)n_in; (void)out_size; (void)ws_size;
  const float* input = (const float*)d_in[0];
  const unsigned char* seqmask = (const unsigned char*)d_in[1];
  const float* noise = (const float*)d_in[3];
  const float* Wih = (const float*)d_in[4];
  const float* Whh = (const float*)d_in[5];
  const float* bih = (const float*)d_in[6];
  const float* bhh = (const float*)d_in[7];
  const float* gamma = (const float*)d_in[8];
  const float* beta  = (const float*)d_in[9];
  const float* Wcp = (const float*)d_in[10]; const float* bcp = (const float*)d_in[11];
  const float* Wff = (const float*)d_in[12]; const float* bff = (const float*)d_in[13];
  const float* Wq  = (const float*)d_in[14]; const float* bq  = (const float*)d_in[15];
  const float* Wkv = (const float*)d_in[16]; const float* bkv = (const float*)d_in[17];
  const float* Wou = (const float*)d_in[18]; const float* bou = (const float*)d_in[19];

  float* out_ctx = (float*)d_out;                       // [8,32,512]
  float* out_p   = out_ctx + (long)B_ * K_ * FF_;       // [8,32,1024]

  // bump allocator on d_ws
  char* ws = (char*)d_ws; size_t off = 0;
  auto alloc = [&](size_t bytes) -> void* {
    void* p = ws + off; off = (off + bytes + 255) & ~(size_t)255; return p;
  };
  const long NBT = (long)B_ * T_;                       // 8192 rows
  __bf16* xbf    = (__bf16*)alloc(NBT * F_ * 2);
  __bf16* WihB   = (__bf16*)alloc(2L * G4_ * F_ * 2);
  __bf16* WhhB   = (__bf16*)alloc(2L * G4_ * H_ * 2);
  float*  xg     = (float*) alloc(NBT * G4_ * 4);
  float*  hs1    = (float*) alloc(NBT * H_ * 4);
  __bf16* hs1b   = (__bf16*)alloc(NBT * H_ * 2);
  float*  hs2    = (float*) alloc(NBT * H_ * 4);
  __bf16* hs2b   = (__bf16*)alloc(NBT * H_ * 2);
  float*  normed = (float*) alloc(NBT * H_ * 4);
  __bf16* normb  = (__bf16*)alloc(NBT * H_ * 2);
  __bf16* csumb  = (__bf16*)alloc((long)B_ * K_ * H_ * 2);
  __bf16* WcpB   = (__bf16*)alloc((long)FF_ * H_ * 2);
  __bf16* WffB   = (__bf16*)alloc((long)FF_ * H_ * 2);
  __bf16* WqB    = (__bf16*)alloc((long)FF_ * FF_ * 2);
  __bf16* WkvB   = (__bf16*)alloc(2L * FF_ * FF_ * 2);
  __bf16* WouB   = (__bf16*)alloc((long)F_ * FF_ * 2);
  float*  piv    = (float*) alloc((long)B_ * K_ * FF_ * 4);
  __bf16* pivb   = (__bf16*)alloc((long)B_ * K_ * FF_ * 2);
  float*  ff     = (float*) alloc(NBT * FF_ * 4);
  __bf16* ffb    = (__bf16*)alloc(NBT * FF_ * 2);
  float*  dp     = (float*) alloc((long)B_ * K_ * T_ * 4);
  float*  eofp   = (float*) alloc((long)B_ * K_ * T_ * 4);
  float*  pws    = (float*) alloc((long)B_ * K_ * T_ * 4);
  float*  qf     = (float*) alloc((long)B_ * K_ * FF_ * 4);
  __bf16* qb     = (__bf16*)alloc((long)B_ * K_ * FF_ * 2);
  float*  kv     = (float*) alloc(NBT * 2L * FF_ * 4);
  __bf16* kvb    = (__bf16*)alloc(NBT * 2L * FF_ * 2);
  float*  sco    = (float*) alloc((long)B_ * HN_ * K_ * T_ * 4);
  __bf16* attb   = (__bf16*)alloc((long)B_ * HN_ * K_ * T_ * 2);
  float*  ctx    = (float*) alloc((long)B_ * K_ * FF_ * 4);
  __bf16* ctxb   = (__bf16*)alloc((long)B_ * K_ * FF_ * 2);

  auto conv = [&](const float* src, __bf16* dst, long n) {
    f2bf_kernel<<<dim3((unsigned)((n + 255) / 256)), dim3(256), 0, stream>>>(src, dst, n);
  };
  // bf16 mirrors of activations-in and all weights
  conv(input, xbf, NBT * F_);
  conv(Wih, WihB, 2L * G4_ * F_);
  conv(Whh, WhhB, 2L * G4_ * H_);
  conv(Wcp, WcpB, (long)FF_ * H_);
  conv(Wff, WffB, (long)FF_ * H_);
  conv(Wq,  WqB,  (long)FF_ * FF_);
  conv(Wkv, WkvB, 2L * FF_ * FF_);
  conv(Wou, WouB, (long)F_ * FF_);

  const size_t scan_lds = (size_t)(B_ * H_ * 4 + B_ * G4_ * 4 + 16 * H_ * 2);  // 96 KB

  // ---- LSTM layer 0: xg = x@Wih0^T + bih0 + bhh0 ; sequential scan ----
  launch_gemm(stream, xbf, F_, 0, 0, WihB, F_, 1, 0, 0, xg, G4_, 0, 0, nullptr,
              bih, bhh, (int)NBT, G4_, F_, 1.0f, 1, 1);
  lstm_scan_kernel<<<1, 1024, scan_lds, stream>>>(xg, WhhB, hs1, hs1b);
  // ---- LSTM layer 1 ----
  launch_gemm(stream, hs1b, H_, 0, 0, WihB + (long)G4_ * F_, H_, 1, 0, 0, xg, G4_, 0, 0, nullptr,
              bih + G4_, bhh + G4_, (int)NBT, G4_, H_, 1.0f, 1, 1);
  lstm_scan_kernel<<<1, 1024, scan_lds, stream>>>(xg, WhhB + (long)G4_ * H_, hs2, hs2b);

  // ---- LayerNorm + chunk summaries ----
  layernorm_kernel<<<dim3((unsigned)((NBT + 7) / 8)), dim3(256), 0, stream>>>(
      hs2, gamma, beta, normed, normb, (int)NBT);
  chunksum_kernel<<<dim3(B_ * K_), dim3(H_), 0, stream>>>(normed, csumb);

  // ---- pivots / future_frames ----
  launch_gemm(stream, csumb, H_, 0, 0, WcpB, H_, 1, 0, 0, piv, FF_, 0, 0, pivb,
              bcp, nullptr, B_ * K_, FF_, H_, 1.0f, 1, 1);
  launch_gemm(stream, normb, H_, 0, 0, WffB, H_, 1, 0, 0, ff, FF_, 0, 0, ffb,
              bff, nullptr, (int)NBT, FF_, H_, 1.0f, 1, 1);

  // ---- dp[b,k,t] = (pivots . ff) / sqrt(FF), per-batch GEMM ----
  launch_gemm(stream, pivb, FF_, (long)K_ * FF_, 0, ffb, FF_, 1, (long)T_ * FF_, 0,
              dp, T_, (long)K_ * T_, 0, nullptr, nullptr, nullptr,
              K_, T_, FF_, 0.04419417382415922f, B_, 1);
  {
    long n = (long)B_ * K_ * T_;
    eof_kernel<<<dim3((unsigned)((n + 255) / 256)), dim3(256), 0, stream>>>(dp, noise, seqmask, eofp);
    futctx_kernel<<<dim3((unsigned)((n + 255) / 256)), dim3(256), 0, stream>>>(eofp, pws, out_p);
  }

  // ---- q / kv projections ----
  launch_gemm(stream, pivb, FF_, 0, 0, WqB, FF_, 1, 0, 0, qf, FF_, 0, 0, qb,
              bq, nullptr, B_ * K_, FF_, FF_, 1.0f, 1, 1);
  launch_gemm(stream, ffb, FF_, 0, 0, WkvB, FF_, 1, 0, 0, kv, 2 * FF_, 0, 0, kvb,
              bkv, nullptr, (int)NBT, 2 * FF_, FF_, 1.0f, 1, 1);

  // ---- scores[b,h,k,t] = q.k / sqrt(hd) : batched over (b,h) ----
  launch_gemm(stream,
              qb, FF_, (long)K_ * FF_, HD_,
              kvb, 2 * FF_, 1, (long)T_ * 2 * FF_, HD_,
              sco, T_, (long)HN_ * K_ * T_, (long)K_ * T_, nullptr, nullptr, nullptr,
              K_, T_, HD_, 0.125f, B_, HN_);
  softmax_w_kernel<<<dim3((B_ * HN_ * K_ + 7) / 8), dim3(256), 0, stream>>>(sco, pws, attb);

  // ---- ctx[b,k,h*64+f] = attn x V (V K-strided: scalar-B path) ----
  launch_gemm(stream,
              attb, T_, (long)HN_ * K_ * T_, (long)K_ * T_,
              kvb + FF_, 1, 2 * FF_, (long)T_ * 2 * FF_, HD_,
              ctx, FF_, (long)K_ * FF_, HD_, ctxb, nullptr, nullptr,
              K_, HD_, T_, 1.0f, B_, HN_);

  // ---- expected_fut_ctx = ctx @ Wout^T + bout -> d_out ----
  launch_gemm(stream, ctxb, FF_, 0, 0, WouB, FF_, 1, 0, 0, out_ctx, F_, 0, 0, nullptr,
              bou, nullptr, B_ * K_, F_, FF_, 1.0f, 1, 1);
}
